// VoxelSetAbstraction_38422777430239
// MI455X (gfx1250) — matrix operand order repaired and checked by reading, not compiled, not run on hardware
//
#include <hip/hip_runtime.h>

typedef __attribute__((ext_vector_type(16))) _Float16 v16h;
typedef __attribute__((ext_vector_type(8)))  _Float16 v8h;
typedef __attribute__((ext_vector_type(8)))  float    v8f;
typedef __attribute__((ext_vector_type(4)))  unsigned int u32x4;
typedef __attribute__((ext_vector_type(8)))  int      i32x8;
typedef __attribute__((ext_vector_type(4)))  int      i32x4;

#define BN_EPS 1e-5f

// ---------------------------------------------------------------------------
// TDM helper: issue a 1-D tensor_load_to_lds of `elems` dwords starting at
// gptr into LDS byte address ldsAddr. tensorElems = valid dwords remaining in
// the source from gptr; reads past it are OOB and return zero (ISA 8.2), which
// zero-fills the tail tile. Descriptor bitfields per CDNA5 ISA:
//   g0: count=1 | lds_addr | global_addr[56:0] | type=2
//   g1: data_size=4B, tensor_dim0, tile_dim0 (1-D: tile_dim1/2 = 0)
// Tracked by TENSORcnt; caller waits with s_wait_tensorcnt.
// ---------------------------------------------------------------------------
__device__ static inline void tdm_load_1d(const void* gptr, unsigned ldsAddr,
                                          unsigned elems, unsigned tensorElems) {
  unsigned long long ga = (unsigned long long)(size_t)gptr;
  u32x4 g0;
  g0.x = 1u;                                            // count=1 (valid user D#)
  g0.y = ldsAddr;                                       // LDS byte address
  g0.z = (unsigned)(ga & 0xFFFFFFFFu);                  // global_addr[31:0]
  g0.w = (unsigned)((ga >> 32) & 0x01FFFFFFu) | (2u << 30); // addr[56:32] | type=2
  i32x8 g1;
  g1[0] = 0x20000;                                      // data_size=2 (4 bytes)
  g1[1] = (int)((tensorElems & 0xFFFFu) << 16);         // tensor_dim0[15:0]
  g1[2] = (int)((tensorElems >> 16) & 0xFFFFu) | (1 << 16); // dim0[31:16]|dim1=1
  g1[3] = (int)(elems << 16);                           // dim1[31:16]=0 | tile_dim0
  g1[4] = 0;                                            // tile_dim1=0, tile_dim2=0
  g1[5] = (int)tensorElems;                             // tensor_dim0_stride[31:0]
  g1[6] = 0;                                            // stride0[47:32], stride1[15:0]
  g1[7] = 0;
  i32x4 gz = {0, 0, 0, 0};
#if __clang_major__ >= 23
  i32x8 gz8 = {0, 0, 0, 0, 0, 0, 0, 0};
  __builtin_amdgcn_tensor_load_to_lds(g0, g1, gz, gz, gz8, 0);
#else
  __builtin_amdgcn_tensor_load_to_lds(g0, g1, gz, gz, 0);
#endif
}

// ---------------------------------------------------------------------------
// Ball query, dual radius (r0 < r1), index-order preserving. One thread per
// keypoint; xyz tiles streamed into LDS by the Tensor Data Mover, double
// buffered so the DMA of tile t+1 overlaps the distance scan of tile t.
// Matches reference: first `nsample` in-radius indices, empty slots filled
// with first index (or 0 if none), empty flag when zero neighbors.
// ---------------------------------------------------------------------------
#define BQ_TILE 512
__global__ void bq_dual_kernel(const float* __restrict__ xyz, int N,
                               const float* __restrict__ kp, int M,
                               float rr0, float rr1, int S0, int S1,
                               int* __restrict__ idx0, int* __restrict__ idx1,
                               unsigned char* __restrict__ e0,
                               unsigned char* __restrict__ e1) {
  __shared__ float sx[2][BQ_TILE * 3];
  int m = blockIdx.x * blockDim.x + threadIdx.x;
  bool act = m < M;
  float px = 0.f, py = 0.f, pz = 0.f;
  if (act) { px = kp[3 * m]; py = kp[3 * m + 1]; pz = kp[3 * m + 2]; }
  int n0 = 0, n1 = 0;
  int nt = (N + BQ_TILE - 1) / BQ_TILE;
  bool issuer = (threadIdx.x < 32);          // wave 0 drives the TDM
  if (issuer) {
    tdm_load_1d(xyz, (unsigned)(size_t)(void*)&sx[0][0],
                BQ_TILE * 3, (unsigned)N * 3u);
  }
  for (int t = 0; t < nt; ++t) {
    int cur = t & 1;
    int base = t * BQ_TILE;
    int cnt = N - base; if (cnt > BQ_TILE) cnt = BQ_TILE;
    if (issuer) __builtin_amdgcn_s_wait_tensorcnt(0);  // tile t resident
    __syncthreads();                                   // publish to all waves
    if (issuer && (t + 1 < nt)) {                      // prefetch tile t+1
      int nbase = base + BQ_TILE;
      tdm_load_1d(xyz + (size_t)nbase * 3,
                  (unsigned)(size_t)(void*)&sx[1 - cur][0],
                  BQ_TILE * 3, (unsigned)(N - nbase) * 3u);
    }
    if (act && (n0 < S0 || n1 < S1)) {
      const float* sp = sx[cur];
      for (int i = 0; i < cnt; ++i) {
        float dx = sp[3 * i] - px, dy = sp[3 * i + 1] - py, dz = sp[3 * i + 2] - pz;
        float d2 = dx * dx + dy * dy + dz * dz;
        if (d2 < rr1) {
          if (n1 < S1) idx1[(size_t)m * S1 + n1++] = base + i;
          if (d2 < rr0 && n0 < S0) idx0[(size_t)m * S0 + n0++] = base + i;
        }
      }
    }
    __syncthreads();   // all lanes done with sx[cur] before it is re-filled
  }
  if (act) {
    int f0 = (n0 > 0) ? idx0[(size_t)m * S0] : 0;
    for (int s = n0; s < S0; ++s) idx0[(size_t)m * S0 + s] = f0;
    int f1 = (n1 > 0) ? idx1[(size_t)m * S1] : 0;
    for (int s = n1; s < S1; ++s) idx1[(size_t)m * S1 + s] = f1;
    e0[m] = (unsigned char)(n0 == 0);
    e1[m] = (unsigned char)(n1 == 0);
  }
}

// ---------------------------------------------------------------------------
// Grouping: row t = (m,s) -> [xyz[idx]-kp[m], feat[idx], 0-pad] as f16.
// ---------------------------------------------------------------------------
__global__ void group_kernel(const float* __restrict__ xyz,
                             const float* __restrict__ feat, int Cf,
                             const float* __restrict__ kp,
                             const int* __restrict__ idx,
                             const unsigned char* __restrict__ empty,
                             _Float16* __restrict__ gmat,
                             int M, int S, int Kpad) {
  int t = blockIdx.x * blockDim.x + threadIdx.x;
  int rows = M * S;
  if (t >= rows) return;
  int m = t / S;
  _Float16* row = gmat + (size_t)t * Kpad;
  if (empty[m]) {
    for (int k = 0; k < Kpad; ++k) row[k] = (_Float16)0.f;
    return;
  }
  int j = idx[t];
  row[0] = (_Float16)(xyz[3 * j + 0] - kp[3 * m + 0]);
  row[1] = (_Float16)(xyz[3 * j + 1] - kp[3 * m + 1]);
  row[2] = (_Float16)(xyz[3 * j + 2] - kp[3 * m + 2]);
  int k = 3;
  for (int c = 0; c < Cf; ++c, ++k) row[k] = (_Float16)feat[(size_t)j * Cf + c];
  for (; k < Kpad; ++k) row[k] = (_Float16)0.f;
}

// Weight prep: W [Cin, Cout] f32 -> Bt [Cout, Kpad] f16 (transposed, 0-pad).
__global__ void wprep_kernel(const float* __restrict__ W, int Cin, int Cout,
                             int Kpad, _Float16* __restrict__ Bt) {
  int t = blockIdx.x * blockDim.x + threadIdx.x;
  if (t >= Cout * Kpad) return;
  int n = t / Kpad, k = t - n * Kpad;
  Bt[t] = (_Float16)(k < Cin ? W[(size_t)k * Cout + n] : 0.f);
}

// ---------------------------------------------------------------------------
// WMMA GEMM: C[rows,Cout] = A[rows,Kpad](f16) * Bt[Cout,Kpad]^T(f16), f32 acc.
// One wave computes a 64x16 strip (4 M-tiles sharing one B fragment) so each
// B load feeds 4 v_wmma_f32_16x16x32_f16. rows must be a multiple of 64.
// Fragment layouts per CDNA5 ISA 7.12.2; all loads are contiguous b128.
// ---------------------------------------------------------------------------
#define GEMM_MT 4
__global__ void gemm_wmma_kernel(const _Float16* __restrict__ A,
                                 const _Float16* __restrict__ Bt,
                                 float* __restrict__ C,
                                 int rows, int Kpad, int Cout) {
  int tilesN = Cout >> 4;
  int tileN  = blockIdx.x % tilesN;
  int tileM0 = (blockIdx.x / tilesN) * GEMM_MT;
  int lane = threadIdx.x & 31;
  int kbA = (lane & 16) ? 8 : 0;    // A: half-wave K interleave
  int cB  = (tileN << 4) + (lane & 15);
  int kbB = (lane & 16) ? 8 : 0;    // B: half-wave K interleave
  const _Float16* ap = A  + (size_t)(tileM0 * 16 + (lane & 15)) * Kpad + kbA;
  const _Float16* bp = Bt + (size_t)cB * Kpad + kbB;
  const size_t mStride = (size_t)16 * Kpad;   // one 16-row M-tile
  v8f acc[GEMM_MT];
#pragma unroll
  for (int t = 0; t < GEMM_MT; ++t) acc[t] = 0.f;
  for (int kc = 0; kc < Kpad; kc += 32) {
    v8h b0 = *(const v8h*)(bp + kc);        // K = kbB .. kbB+7
    v8h b1 = *(const v8h*)(bp + kc + 16);   // K = kbB+16 .. kbB+23
    v16h b;
#pragma unroll
    for (int i = 0; i < 8; ++i) { b[i] = b0[i]; b[i + 8] = b1[i]; }
#pragma unroll
    for (int t = 0; t < GEMM_MT; ++t) {
      v8h a0 = *(const v8h*)(ap + t * mStride + kc);        // K = kbA..kbA+7
      v8h a1 = *(const v8h*)(ap + t * mStride + kc + 16);   // K = kbA+16..+23
      v16h a;
#pragma unroll
      for (int i = 0; i < 8; ++i) { a[i] = a0[i]; a[i + 8] = a1[i]; }
      acc[t] = __builtin_amdgcn_wmma_f32_16x16x32_f16(false, a, false, b,
                                                      (short)0, acc[t],
                                                      false, false);
    }
  }
  int rBase = (lane & 16) ? 8 : 0;
  int col = (tileN << 4) + (lane & 15);
#pragma unroll
  for (int t = 0; t < GEMM_MT; ++t) {
    float* cp = C + (size_t)((tileM0 + t) * 16 + rBase) * Cout + col;
#pragma unroll
    for (int r = 0; r < 8; ++r) cp[(size_t)r * Cout] = acc[t][r];
  }
}

// ---------------------------------------------------------------------------
// Column mean / inv-std over all rows (population var, like jnp.mean/var).
// One block per column; deterministic tree reduction (no float atomics).
// ---------------------------------------------------------------------------
__global__ void colstats_kernel(const float* __restrict__ H, int rows, int C,
                                float* __restrict__ mean,
                                float* __restrict__ invstd) {
  __shared__ float s1[256], s2[256];
  int c = blockIdx.x;
  float a = 0.f, b = 0.f;
  for (int r = threadIdx.x; r < rows; r += blockDim.x) {
    float v = H[(size_t)r * C + c];
    a += v; b += v * v;
  }
  s1[threadIdx.x] = a; s2[threadIdx.x] = b;
  __syncthreads();
  for (int o = blockDim.x >> 1; o > 0; o >>= 1) {
    if ((int)threadIdx.x < o) {
      s1[threadIdx.x] += s1[threadIdx.x + o];
      s2[threadIdx.x] += s2[threadIdx.x + o];
    }
    __syncthreads();
  }
  if (threadIdx.x == 0) {
    float inv = 1.f / (float)rows;
    float mu = s1[0] * inv;
    float var = s2[0] * inv - mu * mu;
    mean[c] = mu;
    invstd[c] = rsqrtf(var + BN_EPS);
  }
}

// BN + ReLU -> f16 activation buffer (K-padded for next WMMA layer).
__global__ void bnrelu_f16_kernel(const float* __restrict__ H, int rows, int C,
                                  int Kpad,
                                  const float* __restrict__ mean,
                                  const float* __restrict__ invstd,
                                  const float* __restrict__ gamma,
                                  const float* __restrict__ beta,
                                  _Float16* __restrict__ out) {
  int t = blockIdx.x * blockDim.x + threadIdx.x;
  if (t >= rows * Kpad) return;
  int r = t / Kpad, c = t - r * Kpad;
  float v = 0.f;
  if (c < C) {
    v = (H[(size_t)r * C + c] - mean[c]) * invstd[c] * gamma[c] + beta[c];
    v = v > 0.f ? v : 0.f;
  }
  out[t] = (_Float16)v;
}

// BN + ReLU + max over S neighbors -> concat feature buffer column slice.
__global__ void bn_maxpool_kernel(const float* __restrict__ H, int M, int S,
                                  int C,
                                  const float* __restrict__ mean,
                                  const float* __restrict__ invstd,
                                  const float* __restrict__ gamma,
                                  const float* __restrict__ beta,
                                  float* __restrict__ F, int Ftot, int colOff) {
  int t = blockIdx.x * blockDim.x + threadIdx.x;
  if (t >= M * C) return;
  int m = t / C, c = t - m * C;
  float mu = mean[c], is = invstd[c], g = gamma[c], b = beta[c];
  const float* hp = H + (size_t)m * S * C + c;
  float best = -3.4e38f;
  for (int s = 0; s < S; ++s) {
    float v = (hp[(size_t)s * C] - mu) * is * g + b;
    v = v > 0.f ? v : 0.f;
    if (v > best) best = v;
  }
  F[(size_t)m * Ftot + colOff + c] = best;
}

// Exact port of bilinear_interpolate_torch on bev [64,64,128].
__global__ void bev_interp_kernel(const float* __restrict__ bev,
                                  const float* __restrict__ kp,
                                  float* __restrict__ F, int M, int Ftot) {
  int t = blockIdx.x * blockDim.x + threadIdx.x;
  if (t >= M * 128) return;
  int m = t / 128, c = t - m * 128;
  float x = (kp[3 * m + 0] + 25.6f) / 0.1f / 8.0f;
  float y = (kp[3 * m + 1] + 25.6f) / 0.1f / 8.0f;
  int x0 = (int)floorf(x), y0 = (int)floorf(y);
  int x1 = x0 + 1, y1 = y0 + 1;
  x0 = x0 < 0 ? 0 : (x0 > 63 ? 63 : x0);
  x1 = x1 < 0 ? 0 : (x1 > 63 ? 63 : x1);
  y0 = y0 < 0 ? 0 : (y0 > 63 ? 63 : y0);
  y1 = y1 < 0 ? 0 : (y1 > 63 ? 63 : y1);
  float wa = ((float)x1 - x) * ((float)y1 - y);
  float wb = ((float)x1 - x) * (y - (float)y0);
  float wc = (x - (float)x0) * ((float)y1 - y);
  float wd = (x - (float)x0) * (y - (float)y0);
  float Ia = bev[((size_t)y0 * 64 + x0) * 128 + c];
  float Ib = bev[((size_t)y1 * 64 + x0) * 128 + c];
  float Ic = bev[((size_t)y0 * 64 + x1) * 128 + c];
  float Id = bev[((size_t)y1 * 64 + x1) * 128 + c];
  F[(size_t)m * Ftot + c] = Ia * wa + Ib * wb + Ic * wc + Id * wd;
}

__global__ void f2h_kernel(const float* __restrict__ src,
                           _Float16* __restrict__ dst, int n) {
  int t = blockIdx.x * blockDim.x + threadIdx.x;
  if (t < n) dst[t] = (_Float16)src[t];
}

__global__ void bnrelu_out_kernel(const float* __restrict__ H, int n, int C,
                                  const float* __restrict__ mean,
                                  const float* __restrict__ invstd,
                                  const float* __restrict__ gamma,
                                  const float* __restrict__ beta,
                                  float* __restrict__ out) {
  int t = blockIdx.x * blockDim.x + threadIdx.x;
  if (t >= n) return;
  int c = t % C;
  float v = (H[t] - mean[c]) * invstd[c] * gamma[c] + beta[c];
  out[t] = v > 0.f ? v : 0.f;
}

// ---------------------------------------------------------------------------
static inline int cdiv_i(int a, int b) { return (a + b - 1) / b; }

extern "C" void kernel_launch(void* const* d_in, const int* in_sizes, int n_in,
                              void* d_out, int out_size, void* d_ws,
                              size_t ws_size, hipStream_t stream) {
  (void)in_sizes; (void)n_in; (void)out_size; (void)ws_size;
  const int M = 2048, FT = 512;
  const float* kp   = (const float*)d_in[0];
  const float* bev  = (const float*)d_in[2];
  const float* fusW = (const float*)d_in[31];
  const float* fusG = (const float*)d_in[32];
  const float* fusB = fusG + 32;

  // -------- workspace carve (regions reused sequentially) --------
  char* base = (char*)d_ws;
  size_t off = 0;
  auto carve = [&](size_t bytes) -> char* {
    off = (off + 255) & ~(size_t)255;
    char* r = base + off;
    off += bytes;
    return r;
  };
  float*         F    = (float*)carve((size_t)M * FT * 4);         // concat [2048,512]
  _Float16*      Ff16 = (_Float16*)carve((size_t)M * FT * 2);
  _Float16*      gmat = (_Float16*)carve((size_t)65536 * 96 * 2);  // grouped A
  float*         hbuf = (float*)carve((size_t)65536 * 64 * 4);     // GEMM out (L0 & L1)
  _Float16*      abuf = (_Float16*)carve((size_t)65536 * 64 * 2);  // activated layer-1 A
  _Float16*      Bt   = (_Float16*)carve((size_t)64 * 512 * 2);    // transposed f16 weights
  int*           idx0 = (int*)carve((size_t)M * 32 * 4);
  int*           idx1 = (int*)carve((size_t)M * 32 * 4);
  unsigned char* e0   = (unsigned char*)carve(M);
  unsigned char* e1   = (unsigned char*)carve(M);
  float*         meanB = (float*)carve(64 * 4);
  float*         invsB = (float*)carve(64 * 4);

  // -------- BEV bilinear interp -> F[:, 0:128] --------
  bev_interp_kernel<<<cdiv_i(M * 128, 256), 256, 0, stream>>>(bev, kp, F, M, FT);

  struct Mod {
    const float* xyz; const float* feat;
    int N, Cf, C0, C1, S0, S1, off0, off1;
    const float* W0; const float* W1; const float* gb0; const float* gb1;
    float r0, r1;
  };
  Mod mods[5] = {
    { (const float*)d_in[1], nullptr,               30000,  0, 16, 16, 16, 16, 128, 144,
      (const float*)d_in[11], (const float*)d_in[12], (const float*)d_in[13], (const float*)d_in[14], 0.4f, 0.8f },
    { (const float*)d_in[3], (const float*)d_in[4], 30000, 16, 16, 16, 16, 16, 160, 176,
      (const float*)d_in[15], (const float*)d_in[16], (const float*)d_in[17], (const float*)d_in[18], 0.4f, 0.8f },
    { (const float*)d_in[5], (const float*)d_in[6], 16000, 32, 32, 32, 16, 32, 192, 224,
      (const float*)d_in[19], (const float*)d_in[20], (const float*)d_in[21], (const float*)d_in[22], 0.8f, 1.2f },
    { (const float*)d_in[7], (const float*)d_in[8],  8000, 64, 64, 64, 16, 32, 256, 320,
      (const float*)d_in[23], (const float*)d_in[24], (const float*)d_in[25], (const float*)d_in[26], 1.2f, 2.4f },
    { (const float*)d_in[9], (const float*)d_in[10], 4000, 64, 64, 64, 16, 32, 384, 448,
      (const float*)d_in[27], (const float*)d_in[28], (const float*)d_in[29], (const float*)d_in[30], 2.4f, 4.8f },
  };

  for (int mi = 0; mi < 5; ++mi) {
    const Mod& md = mods[mi];
    int Cin = 3 + md.Cf;
    int Kp0 = (Cin + 31) & ~31;      // K-pad layer 0
    int Kp1 = (md.C0 + 31) & ~31;    // K-pad layer 1

    bq_dual_kernel<<<cdiv_i(M, 128), 128, 0, stream>>>(
        md.xyz, md.N, kp, M, md.r0 * md.r0, md.r1 * md.r1,
        md.S0, md.S1, idx0, idx1, e0, e1);

    for (int s = 0; s < 2; ++s) {
      int S = s ? md.S1 : md.S0;
      const int* idx = s ? idx1 : idx0;
      const unsigned char* em = s ? e1 : e0;
      int rows = M * S;
      const float* W0s = md.W0 + (size_t)s * Cin * md.C0;
      const float* W1s = md.W1 + (size_t)s * md.C0 * md.C1;
      const float* g0 = md.gb0 + (size_t)s * 2 * md.C0;
      const float* b0 = g0 + md.C0;
      const float* g1 = md.gb1 + (size_t)s * 2 * md.C1;
      const float* b1 = g1 + md.C1;
      int colOff = s ? md.off1 : md.off0;

      // Layer 0: group -> WMMA GEMM -> stats -> BN+ReLU (f16)
      wprep_kernel<<<cdiv_i(md.C0 * Kp0, 256), 256, 0, stream>>>(W0s, Cin, md.C0, Kp0, Bt);
      group_kernel<<<cdiv_i(rows, 256), 256, 0, stream>>>(
          md.xyz, md.feat, md.Cf, kp, idx, em, gmat, M, S, Kp0);
      gemm_wmma_kernel<<<(rows / 64) * (md.C0 / 16), 32, 0, stream>>>(
          gmat, Bt, hbuf, rows, Kp0, md.C0);
      colstats_kernel<<<md.C0, 256, 0, stream>>>(hbuf, rows, md.C0, meanB, invsB);
      bnrelu_f16_kernel<<<cdiv_i(rows * Kp1, 256), 256, 0, stream>>>(
          hbuf, rows, md.C0, Kp1, meanB, invsB, g0, b0, abuf);

      // Layer 1: WMMA GEMM -> stats -> BN+ReLU+maxpool into concat buffer
      wprep_kernel<<<cdiv_i(md.C1 * Kp1, 256), 256, 0, stream>>>(W1s, md.C0, md.C1, Kp1, Bt);
      gemm_wmma_kernel<<<(rows / 64) * (md.C1 / 16), 32, 0, stream>>>(
          abuf, Bt, hbuf, rows, Kp1, md.C1);
      colstats_kernel<<<md.C1, 256, 0, stream>>>(hbuf, rows, md.C1, meanB, invsB);
      bn_maxpool_kernel<<<cdiv_i(M * md.C1, 256), 256, 0, stream>>>(
          hbuf, M, S, md.C1, meanB, invsB, g1, b1, F, FT, colOff);
    }
  }

  // -------- fusion: [2048,512] @ [512,32] + BN1d + ReLU --------
  f2h_kernel<<<cdiv_i(M * FT, 256), 256, 0, stream>>>(F, Ff16, M * FT);
  wprep_kernel<<<cdiv_i(32 * 512, 256), 256, 0, stream>>>(fusW, 512, 32, 512, Bt);
  gemm_wmma_kernel<<<(M / 64) * (32 / 16), 32, 0, stream>>>(Ff16, Bt, hbuf, M, 512, 32);
  colstats_kernel<<<32, 256, 0, stream>>>(hbuf, M, 32, meanB, invsB);
  bnrelu_out_kernel<<<cdiv_i(M * 32, 256), 256, 0, stream>>>(
      hbuf, M * 32, 32, meanB, invsB, fusG, fusB, (float*)d_out);
}
